// MLP_78417512890589
// MI455X (gfx1250) — matrix operand order repaired and checked by reading, not compiled, not run on hardware
//
#include <hip/hip_runtime.h>

// ---------------------------------------------------------------------------
// Spiking MLP block for MI455X (gfx1250, wave32, WMMA + Tensor Data Mover).
// LIF -> GEMM(bf16 WMMA, TDM-staged)+bias -> BN-stats -> BN+LIF -> GEMM -> BN.
// Spikes are exactly {0,1} -> bf16 activation storage is lossless; weights are
// pre-rounded f32->bf16 once; GEMM accumulates in f32 via v_wmma_f32_16x16x32.
// Activations are written channel-last ([row][K]) so every GEMM tile is a
// plain strided 2D bf16 tile -> single tensor_load_to_lds per tile.
// ---------------------------------------------------------------------------

typedef __bf16 bf16_t;
typedef bf16_t v16bf __attribute__((ext_vector_type(16)));
typedef float  v8f   __attribute__((ext_vector_type(8)));
typedef unsigned int u32x4 __attribute__((ext_vector_type(4)));
typedef int          i32x8 __attribute__((ext_vector_type(8)));
typedef int          i32x4 __attribute__((ext_vector_type(4)));

#ifndef __has_builtin
#define __has_builtin(x) 0
#endif
#if __has_builtin(__builtin_amdgcn_tensor_load_to_lds)
#define HAVE_TDM 1
#else
#define HAVE_TDM 0
#endif
#if __has_builtin(__builtin_amdgcn_s_wait_tensorcnt)
#define WAIT_TENSOR(n) __builtin_amdgcn_s_wait_tensorcnt(n)
#else
#define WAIT_TENSOR(n) asm volatile("s_wait_tensorcnt %0" ::"i"(n) : "memory")
#endif

__device__ __forceinline__ unsigned short f2bf(float f) {
    union { float f; unsigned u; } x; x.f = f;
    unsigned r = x.u + 0x7FFFu + ((x.u >> 16) & 1u);   // round-to-nearest-even
    return (unsigned short)(r >> 16);
}

union Frag {
    v16bf v;
    unsigned u[8];
};

#define AS3 __attribute__((address_space(3)))
__device__ __forceinline__ unsigned ldsByteAddr(const void* p) {
    return (unsigned)(unsigned long long)(AS3 const void*)p;
}

// ---------------------------------------------------------------------------
// One-shot weight conversion f32 -> bf16.
// ---------------------------------------------------------------------------
__global__ __launch_bounds__(256) void wcvt_kernel(const float* __restrict__ w,
                                                   unsigned short* __restrict__ wb,
                                                   int n) {
    int i = blockIdx.x * 256 + threadIdx.x;
    if (i < n) wb[i] = f2bf(w[i]);
}

// ---------------------------------------------------------------------------
// Kernel 1: LIF over T on x [T(4),B(32),C(256),HW(256)] f32.
// Spikes stored TRANSPOSED channel-last: s1[(n*256 + p)*256 + c] (bf16).
// ---------------------------------------------------------------------------
__global__ __launch_bounds__(256) void lif1_kernel(const float* __restrict__ x,
                                                   unsigned short* __restrict__ s1) {
    int i = blockIdx.x * 256 + threadIdx.x;   // b*65536 + p*256 + c
    int b = i >> 16;
    int p = (i >> 8) & 255;
    int c = i & 255;
    float v = 0.f;
#pragma unroll
    for (int t = 0; t < 4; ++t) {
        int n = t * 32 + b;
        float xv = x[(((size_t)n) << 16) + (c << 8) + p];
        float h = 0.5f * (v + xv);
        float s = (h >= 1.f) ? 1.f : 0.f;
        v = h * (1.f - s);
        s1[(((size_t)n << 8) + p) * 256 + c] = f2bf(s);   // coalesced in c
    }
}

// ---------------------------------------------------------------------------
// TDM tile load: 128 rows x 32 bf16 (64B rows), global row stride = K elems,
// LDS destination padded to 80B rows (pad 4 DW after every 16 DW).
// D# layout per CDNA5 ISA §8.3/8.4.
// ---------------------------------------------------------------------------
#define LDSK 40   // halfwords per LDS row = 64B data + 16B pad (TDM pad_amount)

#if HAVE_TDM
__device__ __forceinline__ void tdm_load_tile(const unsigned short* gsrc,
                                              unsigned lds_byte, int Kstride) {
    unsigned long long ga = (unsigned long long)gsrc;
    u32x4 g0;
    g0.x = 1u;                                   // count = 1 (valid descriptor)
    g0.y = lds_byte;                             // lds_addr
    g0.z = (unsigned)ga;                         // global_addr[31:0]
    g0.w = (unsigned)(ga >> 32) | (2u << 30);    // global_addr[56:32] | type=2
    i32x8 g1;
    g1[0] = (1 << 16)        // data_size = 2 bytes
          | (1 << 20)        // pad_enable
          | (3 << 22)        // pad_interval code 3 = 16 DWORDs (one 64B row)
          | (3 << 25);       // pad_amount  code 3 = 4 DWORDs (16B)
    g1[1] = (Kstride & 0xFFFF) << 16;            // tensor_dim0[15:0]
    g1[2] = 0;                                   // tensor_dim0 hi, tensor_dim1 lo16
    g1[3] = 1 | (32 << 16);                      // tensor_dim1 = 65536 ; tile_dim0 = 32
    g1[4] = 128;                                 // tile_dim1 = 128 rows ; tile_dim2 = 0
    g1[5] = Kstride;                             // tensor_dim0_stride[31:0]
    g1[6] = 0;                                   // stride hi, dim1_stride lo
    g1[7] = 0;
    i32x4 z4; z4[0] = 0; z4[1] = 0; z4[2] = 0; z4[3] = 0;
    i32x8 z8;
#pragma unroll
    for (int e = 0; e < 8; ++e) z8[e] = 0;
    // 6-arg form (therock-10.0 HIP headers / clang-23 lane).
    __builtin_amdgcn_tensor_load_to_lds(g0, g1, z4, z4, z8, 0);
}
#endif

// ---------------------------------------------------------------------------
// Kernel 2: GEMM  out[nb][o][p] = sum_k A[o][k]*B[nb*256+p][k] + bias[o]
//   A: bf16 [M][K], B: bf16 [128*256 rows][K] (channel-last spikes)
// 256 threads = 8 waves; block tile 128(M)x128(N); wave tile 64x32 (4x2 WMMA).
// K staged 32 wide through double-buffered LDS via TDM (wave 0 issues DMAs).
// ---------------------------------------------------------------------------
__global__ __launch_bounds__(256) void gemm_spike_bf16(
    const unsigned short* __restrict__ A,
    const unsigned short* __restrict__ B,
    const float* __restrict__ bias,
    float* __restrict__ out,
    int M, int K)
{
    __shared__ __align__(16) unsigned short ldsA[2][128 * LDSK]; // [m][k]
    __shared__ __align__(16) unsigned short ldsB[2][128 * LDSK]; // [p][k]

    const int tid  = threadIdx.x;
    const int lane = tid & 31;
    const int wave = tid >> 5;
    const int wm   = wave >> 2;      // 0..1 -> wave M offset *64
    const int wn   = wave & 3;       // 0..3 -> wave N offset *32
    const int lm   = lane & 15;
    const int half = lane >> 4;

    const int ct     = blockIdx.x;        // column tile: 2 per n-block
    const int nb     = ct >> 1;           // time-batch block 0..127
    const int p_base = (ct & 1) * 128;
    const int m_base = blockIdx.y * 128;

    const unsigned short* Ag = A + (size_t)m_base * K;
    const unsigned short* Bg = B + ((size_t)(nb * 256 + p_base)) * K;

    v8f acc[4][2];
#pragma unroll
    for (int i = 0; i < 4; ++i)
#pragma unroll
        for (int j = 0; j < 2; ++j)
#pragma unroll
            for (int e = 0; e < 8; ++e) acc[i][j][e] = 0.f;

    const int nk = K >> 5;

#if HAVE_TDM
    if (wave == 0) {
        tdm_load_tile(Ag, ldsByteAddr(&ldsA[0][0]), K);
        tdm_load_tile(Bg, ldsByteAddr(&ldsB[0][0]), K);
    }
#endif

    for (int kt = 0; kt < nk; ++kt) {
#if HAVE_TDM
        const int buf = kt & 1;
        if (wave == 0) {
            if (kt + 1 < nk) {   // prefetch next tile into other buffer
                tdm_load_tile(Ag + ((kt + 1) << 5), ldsByteAddr(&ldsA[buf ^ 1][0]), K);
                tdm_load_tile(Bg + ((kt + 1) << 5), ldsByteAddr(&ldsB[buf ^ 1][0]), K);
                WAIT_TENSOR(2);  // in-order: current tile's two DMAs complete
            } else {
                WAIT_TENSOR(0);
            }
        }
        __syncthreads();         // current tile visible to all waves
#else
        const int buf = 0;
        __syncthreads();         // previous iteration's reads finished
        const int k0 = kt << 5;
#pragma unroll
        for (int it = 0; it < 2; ++it) {   // 16B chunks: 512 per tile pair
            int chunk = it * 256 + tid;
            int row = chunk >> 2, ko = (chunk & 3) * 8;
            *(uint4*)&ldsA[0][row * LDSK + ko] =
                *(const uint4*)&Ag[(size_t)row * K + k0 + ko];
            *(uint4*)&ldsB[0][row * LDSK + ko] =
                *(const uint4*)&Bg[(size_t)row * K + k0 + ko];
        }
        __syncthreads();
#endif

        // B fragments (32x16, 16-bit): lane n=L%16; lanes 16-31 hold K=16..31.
        Frag bfr[2];
#pragma unroll
        for (int j = 0; j < 2; ++j) {
            const unsigned short* src = &ldsB[buf][(wn * 32 + j * 16 + lm) * LDSK + half * 16];
#pragma unroll
            for (int v = 0; v < 8; ++v)
                bfr[j].u[v] = *(const unsigned*)(src + v * 2);
        }
        // A fragments (16x32, 16-bit): lane m=L%16; lanes 16-31 hold K=16..31.
#pragma unroll
        for (int i = 0; i < 4; ++i) {
            Frag afr;
            const unsigned short* src = &ldsA[buf][(wm * 64 + i * 16 + lm) * LDSK];
#pragma unroll
            for (int v = 0; v < 8; ++v) {
                int kb = half * 16 + (v << 1);
                afr.u[v] = *(const unsigned*)(src + kb);
            }
#pragma unroll
            for (int j = 0; j < 2; ++j) {
                acc[i][j] = __builtin_amdgcn_wmma_f32_16x16x32_bf16(
                    false, afr.v, false, bfr[j].v, (short)0, acc[i][j], false, false);
            }
        }
#if HAVE_TDM
        __syncthreads();         // all waves done reading before buffer reuse
#endif
    }

    // Epilogue: D layout -> m = v + half*8, n = lane%16. Fuse bias.
#pragma unroll
    for (int i = 0; i < 4; ++i) {
#pragma unroll
        for (int j = 0; j < 2; ++j) {
#pragma unroll
            for (int v = 0; v < 8; ++v) {
                int o = m_base + wm * 64 + i * 16 + half * 8 + v;
                int p = p_base + wn * 32 + j * 16 + lm;
                out[(((size_t)nb * M + o) << 8) + p] = acc[i][j][v] + bias[o];
            }
        }
    }
}

// ---------------------------------------------------------------------------
// Kernel 3: per-channel batch stats over 128*256 = 32768 elements.
// stats[ch] = {mean, rsqrt(var + eps)}  (biased var, matches jnp.var).
// ---------------------------------------------------------------------------
__global__ __launch_bounds__(256) void bn_stats_kernel(const float* __restrict__ h,
                                                       float* __restrict__ stats,
                                                       int CHN) {
    int ch = blockIdx.x;
    float s = 0.f, q = 0.f;
    for (int i = threadIdx.x; i < 32768; i += 256) {
        int n = i >> 8, p = i & 255;
        float v = h[(((size_t)n * CHN + ch) << 8) + p];
        s += v; q += v * v;
    }
    __shared__ float rs[256], rq[256];
    rs[threadIdx.x] = s; rq[threadIdx.x] = q;
    __syncthreads();
    for (int d = 128; d > 0; d >>= 1) {
        if (threadIdx.x < d) {
            rs[threadIdx.x] += rs[threadIdx.x + d];
            rq[threadIdx.x] += rq[threadIdx.x + d];
        }
        __syncthreads();
    }
    if (threadIdx.x == 0) {
        float mean = rs[0] * (1.f / 32768.f);
        float var  = rq[0] * (1.f / 32768.f) - mean * mean;
        stats[2 * ch]     = mean;
        stats[2 * ch + 1] = rsqrtf(var + 1e-5f);
    }
}

// ---------------------------------------------------------------------------
// Kernel 4: fused BN (affine) + LIF over T; spikes written channel-last
// s2[(n*256 + p)*CHN + o] (coalesced in o).
// ---------------------------------------------------------------------------
__global__ __launch_bounds__(256) void bn_lif_kernel(const float* __restrict__ h,
                                                     const float* __restrict__ stats,
                                                     const float* __restrict__ g,
                                                     const float* __restrict__ be,
                                                     unsigned short* __restrict__ sp,
                                                     int CHN) {
    int i = blockIdx.x * 256 + threadIdx.x;     // b*(256*CHN) + p*CHN + o
    int o = i % CHN;
    int r = i / CHN;
    int p = r & 255;
    int b = r >> 8;
    float scale = stats[2 * o + 1] * g[o];
    float shift = be[o] - stats[2 * o] * scale;
    float v = 0.f;
#pragma unroll
    for (int t = 0; t < 4; ++t) {
        int n = t * 32 + b;
        float hv = h[(((size_t)n * CHN + o) << 8) + p] * scale + shift;
        float u = 0.5f * (v + hv);
        float s = (u >= 1.f) ? 1.f : 0.f;
        v = u * (1.f - s);
        sp[((size_t)(n * 256 + p)) * CHN + o] = f2bf(s);
    }
}

// ---------------------------------------------------------------------------
// Kernel 5: final BN apply -> output f32 [n][c][p].
// ---------------------------------------------------------------------------
__global__ __launch_bounds__(256) void bn_apply_kernel(const float* __restrict__ ypre,
                                                       const float* __restrict__ stats,
                                                       const float* __restrict__ g,
                                                       const float* __restrict__ be,
                                                       float* __restrict__ out) {
    int i = blockIdx.x * 256 + threadIdx.x;  // n*65536 + c*256 + p
    int c = (i >> 8) & 255;
    float scale = stats[2 * c + 1] * g[c];
    float shift = be[c] - stats[2 * c] * scale;
    out[i] = ypre[i] * scale + shift;
}

// ---------------------------------------------------------------------------
extern "C" void kernel_launch(void* const* d_in, const int* in_sizes, int n_in,
                              void* d_out, int out_size, void* d_ws, size_t ws_size,
                              hipStream_t stream) {
    (void)in_sizes; (void)n_in; (void)out_size; (void)ws_size;
    const float* x   = (const float*)d_in[0];
    // d_in[1] = alpha: only affects surrogate gradient, unused in forward.
    const float* w1  = (const float*)d_in[2];
    const float* b1  = (const float*)d_in[3];
    const float* g1  = (const float*)d_in[4];
    const float* be1 = (const float*)d_in[5];
    const float* w2  = (const float*)d_in[6];
    const float* b2  = (const float*)d_in[7];
    const float* g2  = (const float*)d_in[8];
    const float* be2 = (const float*)d_in[9];
    float* out = (float*)d_out;

    char* ws = (char*)d_ws;
    unsigned short* s1  = (unsigned short*)ws;                        // 16 MiB  spikes1 [row][256]
    unsigned short* s2  = (unsigned short*)(ws + (size_t)16777216);   // 64 MiB  spikes2 [row][1024]
    float*          h   = (float*)(ws + (size_t)83886080);            // 128 MiB hidden f32
    unsigned short* w1b = (unsigned short*)(ws + (size_t)218103808);  // 512 KiB
    unsigned short* w2b = (unsigned short*)(ws + (size_t)218628096);  // 512 KiB
    float* stats1       = (float*)(ws + (size_t)219152384);
    float* stats2       = (float*)(ws + (size_t)219160576);
    float* ypre         = h;  // hidden buffer dead after bn_lif -> reuse

    wcvt_kernel    <<<1024, 256, 0, stream>>>(w1, w1b, 1024 * 256);
    wcvt_kernel    <<<1024, 256, 0, stream>>>(w2, w2b, 256 * 1024);
    lif1_kernel    <<<8192, 256, 0, stream>>>(x, s1);
    gemm_spike_bf16<<<dim3(256, 8), 256, 0, stream>>>(w1b, s1, b1, h, 1024, 256);
    bn_stats_kernel<<<1024, 256, 0, stream>>>(h, stats1, 1024);
    bn_lif_kernel  <<<32768, 256, 0, stream>>>(h, stats1, g1, be1, s2, 1024);
    gemm_spike_bf16<<<dim3(256, 2), 256, 0, stream>>>(w2b, s2, b2, ypre, 256, 1024);
    bn_stats_kernel<<<256, 256, 0, stream>>>(ypre, stats2, 256);
    bn_apply_kernel<<<32768, 256, 0, stream>>>(ypre, stats2, g2, be2, out);
}
